// YoloLoss_60756607369216
// MI455X (gfx1250) — compile-verified, matrix operand order checked by reading, not run on hardware
//
#include <hip/hip_runtime.h>

// YOLO loss, MI455X (gfx1250, wave32).
// Memory-bound reduction: ~177 MB read once -> ~7.6 us at 23.3 TB/s.
// Cross-lane reduction done with V_WMMA_F32_16X16X32_F16 (ones-vector matmul,
// f16 hi/lo split for ~f32 accuracy), then LDS + global float atomics.

typedef __attribute__((ext_vector_type(16))) _Float16 v16h;
typedef __attribute__((ext_vector_type(8)))  float    v8f;

#define INV_S   (1.0f / 14.0f)
#define CELLS_S 14
#define L_COORD 5.0f
#define L_NOOBJ 0.5f

// Sum 4 independent f32 partials across the 32 lanes of a wave using WMMA.
// A(16x32,f16): this lane contributes (hi,lo) split of its value in its two
// A-slots, rest zero. B(32x16) = ones -> D[m][n] = rowsum(A,m) for every n.
// Lanes 0-15 hold D rows 0-7, lanes 16-31 rows 8-15; summing the 8 D regs and
// one xor-16 shuffle gives the full 32-lane total in every lane.
// Requires EXEC all ones: callers must not diverge (no early returns).
__device__ __forceinline__ float4 wave_sum4_wmma(float r0, float r1, float r2, float r3) {
  float in[4] = {r0, r1, r2, r3};
  float out[4];
  v16h b = {};
#pragma unroll
  for (int i = 0; i < 16; ++i) b[i] = (_Float16)1.0f;
#pragma unroll
  for (int j = 0; j < 4; ++j) {
    float v = in[j];
    _Float16 hi = (_Float16)v;
    _Float16 lo = (_Float16)(v - (float)hi);
    v16h a = {};
    a[0] = hi;
    a[1] = lo;
    v8f c = {};
    v8f d = __builtin_amdgcn_wmma_f32_16x16x32_f16(
        /*neg_a=*/false, a, /*neg_b=*/false, b,
        /*c_mod=*/(short)0, c, /*reuse_a=*/false, /*reuse_b=*/false);
    float s = d[0] + d[1] + d[2] + d[3] + d[4] + d[5] + d[6] + d[7];
    s += __shfl_xor(s, 16, 32);   // combine the two row halves -> all lanes
    out[j] = s;
  }
  return make_float4(out[0], out[1], out[2], out[3]);
}

__global__ void yolo_zero_acc(float* __restrict__ acc) {
  if (threadIdx.x < 4) acc[threadIdx.x] = 0.0f;
}

__global__ __launch_bounds__(256) void yolo_main(
    const float* __restrict__ pred,   // (N,14,14,30)
    const float* __restrict__ tbox,   // (N,14,14,4)
    const float* __restrict__ tcls,   // (N,14,14,20)
    const int*   __restrict__ mask,   // (N,14,14) bool->int32
    float* __restrict__ acc,          // [reg, contain, noobj, cls]
    int ncells) {
  __shared__ float sAcc[4];
  const int tid = threadIdx.x;
  if (tid < 4) sAcc[tid] = 0.0f;
  __syncthreads();

  const int t = blockIdx.x * blockDim.x + tid;
  const bool valid = (t < ncells);
  const int tt = valid ? t : 0;          // tail lanes read cell 0, weight 0

  // ---- loads (coalesced, vectorized) ----
  float p[30];
  {
    const float2* pp2 = (const float2*)(pred + (size_t)tt * 30);  // 8B aligned
#pragma unroll
    for (int k = 0; k < 15; ++k) {
      float2 v = pp2[k];
      p[2 * k] = v.x;
      p[2 * k + 1] = v.y;
    }
  }
  const float4 tb = *(const float4*)(tbox + (size_t)tt * 4);      // 16B aligned
  float tcv[20];
  {
    const float4* tc4 = (const float4*)(tcls + (size_t)tt * 20);  // 16B aligned
#pragma unroll
    for (int k = 0; k < 5; ++k) {
      float4 v = tc4[k];
      tcv[4 * k] = v.x; tcv[4 * k + 1] = v.y;
      tcv[4 * k + 2] = v.z; tcv[4 * k + 3] = v.w;
    }
  }
  const bool m = (mask[tt] != 0);

  // ---- class loss ----
  float cls = 0.0f;
#pragma unroll
  for (int k = 0; k < 20; ++k) {
    float d = p[10 + k] - tcv[k];
    cls = fmaf(d, d, cls);
  }
  cls = m ? cls : 0.0f;

  // ---- no-object confidence loss ----
  const float c0 = p[4], c1 = p[9];
  const float noobj = m ? 0.0f : L_NOOBJ * (c0 * c0 + c1 * c1);

  // ---- xywh -> "xyxy" (reference's exact formula) ----
  float T0 = tb.x * INV_S - 0.5f * tb.z;
  float T1 = tb.y * INV_S - 0.5f * tb.w;
  float T2 = tb.z * INV_S + 0.5f * tb.z;
  float T3 = tb.w * INV_S + 0.5f * tb.w;

  float A0 = p[0] * INV_S - 0.5f * p[2];
  float A1 = p[1] * INV_S - 0.5f * p[3];
  float A2 = p[2] * INV_S + 0.5f * p[2];
  float A3 = p[3] * INV_S + 0.5f * p[3];

  float B0 = p[5] * INV_S - 0.5f * p[7];
  float B1 = p[6] * INV_S - 0.5f * p[8];
  float B2 = p[7] * INV_S + 0.5f * p[7];
  float B3 = p[8] * INV_S + 0.5f * p[8];

  const float tArea = (T2 - T0) * (T3 - T1);

  // IoU box A
  float w = fmaxf(fminf(A2, T2) - fmaxf(A0, T0), 0.0f);
  float h = fmaxf(fminf(A3, T3) - fmaxf(A1, T1), 0.0f);
  float inter = w * h;
  float iou0 = inter / ((A2 - A0) * (A3 - A1) + tArea - inter);
  // IoU box B
  w = fmaxf(fminf(B2, T2) - fmaxf(B0, T0), 0.0f);
  h = fmaxf(fminf(B3, T3) - fmaxf(B1, T1), 0.0f);
  inter = w * h;
  float iou1 = inter / ((B2 - B0) * (B3 - B1) + tArea - inter);

  // argmax (first max on tie, like jnp.argmax)
  const bool pick1 = (iou1 > iou0);
  const float bx0 = pick1 ? B0 : A0;
  const float bx1 = pick1 ? B1 : A1;
  const float bx2 = pick1 ? B2 : A2;
  const float bx3 = pick1 ? B3 : A3;
  const float bconf = pick1 ? c1 : c0;
  const float biou  = pick1 ? iou1 : iou0;

  const float dx = T0 - bx0, dy = T1 - bx1;
  const float sw = sqrtf(T2) - sqrtf(bx2);
  const float sh = sqrtf(T3) - sqrtf(bx3);
  const float reg = m ? L_COORD * (dx * dx + dy * dy + sw * sw + sh * sh) : 0.0f;
  const float dcf = bconf - biou;
  const float contain = m ? dcf * dcf : 0.0f;

  // ---- reduction (EXEC all ones; tail weighted to zero) ----
  const float vf = valid ? 1.0f : 0.0f;
  const float4 tot = wave_sum4_wmma(reg * vf, contain * vf, noobj * vf, cls * vf);

  const int lane = tid & 31;
  if (lane < 4) {
    const float v = (lane == 0) ? tot.x : (lane == 1) ? tot.y
                  : (lane == 2) ? tot.z : tot.w;
    atomicAdd(&sAcc[lane], v);   // ds_add_f32
  }
  __syncthreads();
  if (tid < 4) atomicAdd(&acc[tid], sAcc[tid]);
}

__global__ void yolo_finalize(const float* __restrict__ acc,
                              float* __restrict__ out, float invN) {
  if (threadIdx.x == 0) {
    const float reg = acc[0], con = acc[1], no = acc[2], cls = acc[3];
    const float total = cls + no + reg + con;
    out[0] = total * invN;
    out[1] = reg * invN;
    out[2] = con * invN;
    out[3] = no * invN;
    out[4] = cls * invN;
  }
}

extern "C" void kernel_launch(void* const* d_in, const int* in_sizes, int n_in,
                              void* d_out, int out_size, void* d_ws, size_t ws_size,
                              hipStream_t stream) {
  const float* pred = (const float*)d_in[0];
  const float* tbox = (const float*)d_in[1];
  const float* tcls = (const float*)d_in[2];
  const int*   mask = (const int*)d_in[3];    // bool map uploaded as int32
  float* acc = (float*)d_ws;                  // 4 f32 accumulators
  float* out = (float*)d_out;                 // 5 f32 outputs

  const int ncells = in_sizes[3];             // N * 14 * 14
  const int N = ncells / (CELLS_S * CELLS_S);
  const int blocks = (ncells + 255) / 256;

  yolo_zero_acc<<<1, 32, 0, stream>>>(acc);   // re-zero every call (graph-safe)
  yolo_main<<<blocks, 256, 0, stream>>>(pred, tbox, tcls, mask, acc, ncells);
  yolo_finalize<<<1, 32, 0, stream>>>(acc, out, 1.0f / (float)N);
}
